// AGCN_32873679684362
// MI455X (gfx1250) — matrix operand order repaired and checked by reading, not compiled, not run on hardware
//
#include <hip/hip_runtime.h>

// ---------------------------------------------------------------------------
// AGCN forward for MI455X (gfx1250, wave32, WMMA).
// All GEMM-shaped work runs on V_WMMA_F32_16X16X32_BF16 (f32 accumulate,
// bf16 operands converted in-register from fp32 activations/weights).
// ---------------------------------------------------------------------------

typedef __attribute__((ext_vector_type(16))) __bf16 v16bf;
typedef __attribute__((ext_vector_type(8)))  float  v8f;

#define GF_BN  1
#define GF_ACC 2

#ifndef BN_EPS
#define BN_EPS 1e-5f
#endif

// ---------------------------------------------------------------------------
// Generic convolution-as-GEMM with WMMA.  V (joints) fixed at 25.
//   Out[n, o, t, v] = epi( sum_{c,k} W[o, c*KW+k] * X[n, c, t*stride+k-pad, v] )
// KW=1: all 1x1 convs (incl. strided residual convs; pad==0, and for any
//        stride the sampled t*stride is provably in range) ;  KW=9: TCN.
// Block = 128 threads = 4 waves; each wave owns a 16(O) x 16(M=T*25) tile.
// Guarded paths use clamp-address + value-select (no EXEC divergence);
// full tiles take unguarded vectorizable loads.  32-bit indices throughout.
// ---------------------------------------------------------------------------
template <int KW>
__global__ __launch_bounds__(128) void gemm_conv_kernel(
    const float* __restrict__ X, const float* __restrict__ W,
    const float* __restrict__ bias,
    const float* __restrict__ bn_g, const float* __restrict__ bn_b,
    const float* __restrict__ bn_m, const float* __restrict__ bn_v,
    float* __restrict__ Out,
    int O, int Cin, int Tin, int Tout,
    int stride, int pad, int flags)
{
  const int M    = Tout * 25;
  const int Ktot = Cin * KW;
  __shared__ float lds[4][32 * 16];

  const int wv     = threadIdx.x >> 5;
  const int lane   = threadIdx.x & 31;
  const int laneHi = lane >> 4;
  const int l16    = lane & 15;
  const int n      = blockIdx.z;
  const int o0     = blockIdx.y * 16;
  const int m0     = (blockIdx.x * 4 + wv) * 16;

  // wave-uniform decomposition of m0 (constant divisor 25)
  const int t_base = m0 / 25;
  const int v_base = m0 - t_base * 25;
  const bool tileOk = (m0 + 16 <= M);

  const float* __restrict__ Xn = X + (size_t)n * Cin * Tin * 25;
  float* __restrict__ On       = Out + (size_t)n * O * M;

  v8f acc = {0.f, 0.f, 0.f, 0.f, 0.f, 0.f, 0.f, 0.f};
  const int ksteps = (Ktot + 31) >> 5;

  for (int ks = 0; ks < ksteps; ++ks) {
    const int k0 = ks << 5;
    // ---- stage B tile (rows kk=k0..k0+31, cols m0..m0+15) into LDS ----
    {
      const int kk = k0 + lane;
      int c         = (KW == 1) ? kk : (kk / KW);     // constant divisor
      const int kwi = (KW == 1) ? 0  : (kk - c * KW);
      const bool krow_ok = (kk < Ktot);
      if (c > Cin - 1) c = Cin - 1;                   // clamp: safe spec loads
      const unsigned cBase = (unsigned)c * Tin * 25;
      float* row = &lds[wv][lane * 16];
      if (krow_ok & tileOk) {
        if (KW == 1) {
          // t*stride always in [0,Tin) for t<Tout (stride 1 or 2, pad 0)
#pragma unroll
          for (int j = 0; j < 16; ++j) {
            int vv = v_base + j, t = t_base;
            if (vv >= 25) { vv -= 25; t += 1; }
            row[j] = Xn[cBase + (unsigned)(t * stride) * 25 + vv];
          }
        } else {
#pragma unroll
          for (int j = 0; j < 16; ++j) {
            int vv = v_base + j, t = t_base;
            if (vv >= 25) { vv -= 25; t += 1; }
            const int ti  = t * stride + kwi - pad;
            const int tiC = ti < 0 ? 0 : (ti >= Tin ? Tin - 1 : ti);
            const float f = Xn[cBase + (unsigned)tiC * 25 + vv];
            row[j] = (ti >= 0 && ti < Tin) ? f : 0.f;
          }
        }
      } else {
#pragma unroll
        for (int j = 0; j < 16; ++j) {
          int vv = v_base + j, t = t_base;
          if (vv >= 25) { vv -= 25; t += 1; }
          const int ti  = t * stride + kwi - pad;
          const int tiC = ti < 0 ? 0 : (ti >= Tin ? Tin - 1 : ti);
          const float f = Xn[cBase + (unsigned)tiC * 25 + vv];
          const bool ok = krow_ok && (m0 + j < M) && ti >= 0 && ti < Tin;
          row[j] = ok ? f : 0.f;
        }
      }
    }
    __syncthreads();

    // ---- A fragment: weights, row = o0+(lane&15), bf16 in-register ----
    v16bf a;
    {
      const unsigned wBase = (unsigned)(o0 + l16) * Ktot;
      const int kb1 = k0 + (laneHi ? 8 : 0);
      const int kb2 = kb1 + 16;
      if (k0 + 32 <= Ktot) {
#pragma unroll
        for (int j = 0; j < 8; ++j) {
          a[j]     = (__bf16)W[wBase + kb1 + j];
          a[8 + j] = (__bf16)W[wBase + kb2 + j];
        }
      } else {
#pragma unroll
        for (int j = 0; j < 8; ++j) {
          const int i1 = kb1 + j, i2 = kb2 + j;
          const float f1 = W[wBase + (i1 < Ktot ? i1 : Ktot - 1)];
          const float f2 = W[wBase + (i2 < Ktot ? i2 : Ktot - 1)];
          a[j]     = (__bf16)((i1 < Ktot) ? f1 : 0.f);
          a[8 + j] = (__bf16)((i2 < Ktot) ? f2 : 0.f);
        }
      }
    }
    // ---- B fragment from LDS: col = lane&15, K = (laneHi?8:0)+e+(e>=8?8:0) ----
    v16bf b;
    {
      const int rb = laneHi ? 8 : 0;
#pragma unroll
      for (int e = 0; e < 16; ++e)
        b[e] = (__bf16)lds[wv][(rb + e + (e >= 8 ? 8 : 0)) * 16 + l16];
    }
    acc = __builtin_amdgcn_wmma_f32_16x16x32_bf16(false, a, false, b,
                                                  (short)0, acc, false, false);
    __syncthreads();
  }

  // ---- epilogue: single m-guard; 32-bit index marched by M per row ----
  const int m  = m0 + l16;
  const int oB = o0 + (laneHi ? 8 : 0);
  if (m < M) {
    unsigned idx = (unsigned)oB * M + m;
    if (flags & GF_BN) {
#pragma unroll
      for (int g = 0; g < 8; ++g) {
        const int o = oB + g;
        float val = acc[g] + bias[o];
        val = (val - bn_m[o]) * (bn_g[o] * rsqrtf(bn_v[o] + BN_EPS)) + bn_b[o];
        On[idx] = val;
        idx += M;
      }
    } else if (flags & GF_ACC) {
#pragma unroll
      for (int g = 0; g < 8; ++g) {
        On[idx] += acc[g] + bias[oB + g];
        idx += M;
      }
    } else {
#pragma unroll
      for (int g = 0; g < 8; ++g) {
        On[idx] = acc[g] + bias[oB + g];
        idx += M;
      }
    }
  }
}

// ---------------------------------------------------------------------------
// xa[n, ct, v] = sum_u x[n, ct, u] * att[n, u, v]   (K = 25, padded to 32)
// One WMMA per 16-row tile; B fragment (att) loaded once per wave.
// A fragment loads are fully unguarded (row clamped; u>=25 columns are
// annihilated by zeroed B rows); stores take an 8x immediate-offset fast path.
// ---------------------------------------------------------------------------
__global__ __launch_bounds__(128) void xa_wmma_kernel(
    const float* __restrict__ X, const float* __restrict__ Att,
    float* __restrict__ Xa, int CT)
{
  const int wv     = threadIdx.x >> 5;
  const int lane   = threadIdx.x & 31;
  const int laneHi = lane >> 4;
  const int l16    = lane & 15;
  const int n      = blockIdx.z;
  const int v0     = blockIdx.y * 16;
  const int ct0    = (blockIdx.x * 4 + wv) * 16;

  const float* __restrict__ An = Att + (size_t)n * 625;
  const float* __restrict__ Xn = X  + (size_t)n * CT * 25;
  float* __restrict__ Yn       = Xa + (size_t)n * CT * 25;

  // B fragment: unconditional clamped loads, zero-select for u >= 25
  v16bf b;
  {
    const int v  = v0 + l16;
    const int vC = v < 25 ? v : 24;
    const int ub = laneHi ? 8 : 0;
#pragma unroll
    for (int e = 0; e < 16; ++e) {
      const int u  = ub + e + (e >= 8 ? 8 : 0);
      const int uC = u < 25 ? u : 24;
      const float f = An[(unsigned)uC * 25 + vC];
      b[e] = (__bf16)((u < 25) ? f : 0.f);
    }
  }
  // A fragment: row clamped; no u guard needed (B rows >=25 are zero)
  v16bf a;
  {
    int row = ct0 + l16;
    if (row > CT - 1) row = CT - 1;
    const unsigned xBase = (unsigned)row * 25;
    const int kb1 = laneHi ? 8 : 0;
    const int kb2 = kb1 + 16;
#pragma unroll
    for (int j = 0; j < 8; ++j) {
      a[j]     = (__bf16)Xn[xBase + kb1 + j];
      a[8 + j] = (__bf16)Xn[xBase + kb2 + j];   // <=7-elt overread: in-ws slack
    }
  }
  v8f acc = {0.f, 0.f, 0.f, 0.f, 0.f, 0.f, 0.f, 0.f};
  acc = __builtin_amdgcn_wmma_f32_16x16x32_bf16(false, a, false, b,
                                                (short)0, acc, false, false);
  // store: single v-guard; immediate-offset fast path for full row tiles
  const int v    = v0 + l16;
  const int rowB = ct0 + (laneHi ? 8 : 0);
  if (v < 25) {
    float* outp = Yn + (unsigned)rowB * 25 + v;
    if (rowB + 8 <= CT) {
#pragma unroll
      for (int g = 0; g < 8; ++g) outp[g * 25] = acc[g];
    } else {
#pragma unroll
      for (int g = 0; g < 8; ++g)
        if (rowB + g < CT) outp[g * 25] = acc[g];
    }
  }
}

// ---------------------------------------------------------------------------
// Attention logits: L[n,u,v] = scale * sum_kk a[n,kk,u] * b[n,kk,v]
// (kk = c*T+t flattening; tiny output, reduction-heavy -> plain VALU)
// ---------------------------------------------------------------------------
__global__ void attn_logits_kernel(const float* __restrict__ A1,
                                   const float* __restrict__ B1,
                                   float* __restrict__ L, int K, float scale)
{
  const int u = blockIdx.x;      // 0..24
  const int n = blockIdx.y;      // 0..63
  const int v = threadIdx.x;     // block of 32
  if (v >= 25) return;
  const float* __restrict__ an = A1 + (size_t)n * K * 25;
  const float* __restrict__ bn_ = B1 + (size_t)n * K * 25;
  float s = 0.f;
  for (int kk = 0; kk < K; ++kk)
    s += an[(unsigned)kk * 25 + u] * bn_[(unsigned)kk * 25 + v];
  L[((unsigned)n * 25 + u) * 25 + v] = s * scale;
}

// softmax over u (rows) per column v, + (A + graph_attn)
__global__ void attn_softmax_kernel(const float* __restrict__ L,
                                    const float* __restrict__ Abase,
                                    const float* __restrict__ Gattn,
                                    float* __restrict__ Att)
{
  const int n = blockIdx.x;
  const int v = threadIdx.x;
  if (v >= 25) return;
  const float* __restrict__ Ln = L + (unsigned)n * 625;
  float mx = -1e30f;
  for (int u = 0; u < 25; ++u) mx = fmaxf(mx, Ln[u * 25 + v]);
  float s = 0.f;
  for (int u = 0; u < 25; ++u) s += expf(Ln[u * 25 + v] - mx);
  const float inv = 1.f / s;
  for (int u = 0; u < 25; ++u)
    Att[(unsigned)n * 625 + u * 25 + v] =
        expf(Ln[u * 25 + v] - mx) * inv + Abase[u * 25 + v] + Gattn[u * 25 + v];
}

// H[i] = relu( bn(H[i]) + R[i] )
// grid: (ceil(M/256), C, N)  ->  no per-element division for channel index
__global__ void bn_add_relu_kernel(float* __restrict__ H,
                                   const float* __restrict__ R,
                                   const float* __restrict__ g,
                                   const float* __restrict__ b,
                                   const float* __restrict__ m,
                                   const float* __restrict__ v,
                                   int M)
{
  const int j = blockIdx.x * blockDim.x + threadIdx.x;
  if (j >= M) return;
  const int o = blockIdx.y;
  const unsigned i = ((unsigned)blockIdx.z * gridDim.y + o) * (unsigned)M + j;
  float val = (H[i] - m[o]) * (g[o] * rsqrtf(v[o] + BN_EPS)) + b[o];
  val += R[i];
  H[i] = fmaxf(val, 0.f);
}

// Y[i] = relu(Y[i] + (R ? R[i] : 0))
__global__ void add_relu_kernel(float* __restrict__ Y,
                                const float* __restrict__ R, unsigned total)
{
  const unsigned i = blockIdx.x * blockDim.x + threadIdx.x;
  if (i >= total) return;
  float v = Y[i] + (R ? R[i] : 0.f);
  Y[i] = fmaxf(v, 0.f);
}

// data_bn + permute: Y[(n*2+m), c, t, v] = bn_{ch}(x[n,c,t,v,m]),
// ch = (m*25+v)*3+c   (N=32,C=3,T=300,V=25,M=2)
__global__ void data_bn_kernel(const float* __restrict__ X,
                               const float* __restrict__ beta,
                               const float* __restrict__ gamma,
                               const float* __restrict__ mean,
                               const float* __restrict__ var,
                               float* __restrict__ Y)
{
  const int total = 64 * 3 * 300 * 25;
  const int i = blockIdx.x * blockDim.x + threadIdx.x;
  if (i >= total) return;
  const int v  = i % 25;
  const int t  = (i / 25) % 300;
  const int c  = (i / (25 * 300)) % 3;
  const int nm = i / (25 * 300 * 3);
  const int n  = nm >> 1, m = nm & 1;
  const float x  = X[((((unsigned)n * 3 + c) * 300 + t) * 25 + v) * 2 + m];
  const int   ch = (m * 25 + v) * 3 + c;
  Y[i] = (x - mean[ch]) * (gamma[ch] * rsqrtf(var[ch] + BN_EPS)) + beta[ch];
}

// pooled[n,c] = mean over m(2) and T*V of Y[(n*2+m), c, :, :]
__global__ void pool_kernel(const float* __restrict__ Y, float* __restrict__ P,
                            int C, int TV)
{
  const int nc = blockIdx.x;
  const int n = nc / C, c = nc % C;
  float s = 0.f;
  for (int m = 0; m < 2; ++m) {
    const float* __restrict__ base = Y + ((size_t)(n * 2 + m) * C + c) * TV;
    for (int j = threadIdx.x; j < TV; j += blockDim.x) s += base[j];
  }
  __shared__ float red[128];
  red[threadIdx.x] = s;
  __syncthreads();
  for (int st = 64; st > 0; st >>= 1) {
    if ((int)threadIdx.x < st) red[threadIdx.x] += red[threadIdx.x + st];
    __syncthreads();
  }
  if (threadIdx.x == 0) P[nc] = red[0] / (2.f * (float)TV);
}

// out[n,k] = fc_b[k] + sum_c pooled[n,c] * fc_w[k,c]
__global__ void fc_kernel(const float* __restrict__ P,
                          const float* __restrict__ Wf,
                          const float* __restrict__ Bf,
                          float* __restrict__ Out, int C)
{
  const int n = blockIdx.x;
  const int k = threadIdx.x;
  if (k >= 60) return;
  float s = Bf[k];
  for (int c = 0; c < C; ++c) s += P[(unsigned)n * C + c] * Wf[(unsigned)k * C + c];
  Out[n * 60 + k] = s;
}

// ---------------------------------------------------------------------------
// Host orchestration
// ---------------------------------------------------------------------------
struct ConvP { const float* b; const float* w; };
struct BnP   { const float* beta; const float* gamma; const float* mean; const float* var; };

extern "C" void kernel_launch(void* const* d_in, const int* in_sizes, int n_in,
                              void* d_out, int out_size, void* d_ws, size_t ws_size,
                              hipStream_t stream)
{
  (void)in_sizes; (void)n_in; (void)out_size; (void)ws_size;

  const float* x = (const float*)d_in[0];
  const float* A = (const float*)d_in[1];
  int p = 2;
  auto nxt = [&]() { return (const float*)d_in[p++]; };

  // params flattened in JAX tree order (dict keys sorted alphabetically)
  const float* db_beta = nxt(); const float* db_gamma = nxt();
  const float* db_mean = nxt(); const float* db_var   = nxt();
  const float* fc_b = nxt();    const float* fc_w = nxt();

  struct Cfg { int cin, cout, stride; bool res; };
  const Cfg cfg[10] = {{3,64,1,false},{64,64,1,true},{64,64,1,true},{64,64,1,true},
                       {64,128,2,true},{128,128,1,true},{128,128,1,true},
                       {128,256,2,true},{256,256,1,true},{256,256,1,true}};

  struct Layer {
    ConvP a[3], b[3], g[3]; BnP gbn; const float* gattn;
    bool hasRes; BnP resBn; ConvP resConv;
    bool hasResTcn; ConvP resTcn; BnP resTcnBn;
    ConvP tcn; BnP tcnBn;
  } L[10];

  for (int i = 0; i < 10; ++i) {
    const Cfg& c = cfg[i];
    Layer& l = L[i];
    // gcn: keys a < b < bn < g < graph_attn < res
    for (int j = 0; j < 3; ++j) { l.a[j].b = nxt(); l.a[j].w = nxt(); }
    for (int j = 0; j < 3; ++j) { l.b[j].b = nxt(); l.b[j].w = nxt(); }
    l.gbn.beta = nxt(); l.gbn.gamma = nxt(); l.gbn.mean = nxt(); l.gbn.var = nxt();
    for (int j = 0; j < 3; ++j) { l.g[j].b = nxt(); l.g[j].w = nxt(); }
    l.gattn = nxt();
    l.hasRes = (c.cin != c.cout);
    if (l.hasRes) { // res: bn < conv
      l.resBn.beta = nxt(); l.resBn.gamma = nxt(); l.resBn.mean = nxt(); l.resBn.var = nxt();
      l.resConv.b = nxt();  l.resConv.w = nxt();
    }
    // layer: gcn < res_tcn < tcn
    l.hasResTcn = c.res && !(c.cin == c.cout && c.stride == 1);
    if (l.hasResTcn) { // res_tcn: b < bn < w
      l.resTcn.b = nxt();
      l.resTcnBn.beta = nxt(); l.resTcnBn.gamma = nxt(); l.resTcnBn.mean = nxt(); l.resTcnBn.var = nxt();
      l.resTcn.w = nxt();
    }
    // tcn: b < bn < w
    l.tcn.b = nxt();
    l.tcnBn.beta = nxt(); l.tcnBn.gamma = nxt(); l.tcnBn.mean = nxt(); l.tcnBn.var = nxt();
    l.tcn.w = nxt();
  }

  // ---- workspace carve-up (floats) ----
  float* ws = (float*)d_ws;
  const size_t SX = 30720000;  // max activation  [64,64,300,25] / [64,256,75,25]
  const size_t SH = 61440000;  // max hidden/res  [64,128,300,25]
  float* bufX[2]  = { ws, ws + SX };
  float* bufH     = ws + 2 * SX;
  float* bufS     = bufH + SH;                // shared a|b|xa, reused as res
  float* bufA     = bufS;
  float* bufB     = bufS + 15360000;
  float* bufXa    = bufS + 30720000;
  float* bufR     = bufS;                     // residual (after a/b/xa retire)
  float* logitsBuf = bufS + SH;               // 64*625
  float* attBuf    = logitsBuf + 40000;       // 64*625
  float* poolBuf   = attBuf + 40000;          // 32*256

  // ---- data_bn + layout permute ----
  {
    const int total = 64 * 3 * 300 * 25;
    data_bn_kernel<<<(total + 255) / 256, 256, 0, stream>>>(
        x, db_beta, db_gamma, db_mean, db_var, bufX[0]);
  }

  int cur = 0;
  int T = 300;
  for (int li = 0; li < 10; ++li) {
    const Cfg& c = cfg[li];
    Layer& l = L[li];
    const int cin = c.cin, cout = c.cout, s = c.stride;
    const int ic = cout / 4;
    const int M  = T * 25;
    float* Xc = bufX[cur];

    // ---------------- GCN: 3 attention branches -> bufH ----------------
    for (int br = 0; br < 3; ++br) {
      dim3 gAB((M + 63) / 64, ic / 16, 64);
      gemm_conv_kernel<1><<<gAB, 128, 0, stream>>>(
          Xc, l.a[br].w, l.a[br].b, nullptr, nullptr, nullptr, nullptr,
          bufA, ic, cin, T, T, 1, 0, 0);
      gemm_conv_kernel<1><<<gAB, 128, 0, stream>>>(
          Xc, l.b[br].w, l.b[br].b, nullptr, nullptr, nullptr, nullptr,
          bufB, ic, cin, T, T, 1, 0, 0);

      const int K = ic * T;
      attn_logits_kernel<<<dim3(25, 64), 32, 0, stream>>>(
          bufA, bufB, logitsBuf, K, 1.f / (float)K);
      attn_softmax_kernel<<<64, 32, 0, stream>>>(
          logitsBuf, A + br * 625, l.gattn + br * 625, attBuf);

      const int CT = cin * T;
      dim3 gXa((((CT + 15) / 16) + 3) / 4, 2, 64);
      xa_wmma_kernel<<<gXa, 128, 0, stream>>>(Xc, attBuf, bufXa, CT);

      dim3 gG((M + 63) / 64, cout / 16, 64);
      gemm_conv_kernel<1><<<gG, 128, 0, stream>>>(
          bufXa, l.g[br].w, l.g[br].b, nullptr, nullptr, nullptr, nullptr,
          bufH, cout, cin, T, T, 1, 0, br ? GF_ACC : 0);
    }

    // gcn residual (bn(conv(x)) when cin!=cout, else identity)
    const float* Rg = Xc;
    if (l.hasRes) {
      dim3 gR((M + 63) / 64, cout / 16, 64);
      gemm_conv_kernel<1><<<gR, 128, 0, stream>>>(
          Xc, l.resConv.w, l.resConv.b,
          l.resBn.gamma, l.resBn.beta, l.resBn.mean, l.resBn.var,
          bufR, cout, cin, T, T, 1, 0, GF_BN);
      Rg = bufR;
    }
    {
      dim3 gBN((M + 255) / 256, cout, 64);
      bn_add_relu_kernel<<<gBN, 256, 0, stream>>>(
          bufH, Rg, l.gbn.gamma, l.gbn.beta, l.gbn.mean, l.gbn.var, M);
    }

    // ---------------- TCN (9-tap, stride s) ----------------
    const int Tout = (T - 1) / s + 1;
    const int Mo   = Tout * 25;
    float* Yn = bufX[1 - cur];
    dim3 gT((Mo + 63) / 64, cout / 16, 64);
    gemm_conv_kernel<9><<<gT, 128, 0, stream>>>(
        bufH, l.tcn.w, l.tcn.b,
        l.tcnBn.gamma, l.tcnBn.beta, l.tcnBn.mean, l.tcnBn.var,
        Yn, cout, cout, T, Tout, s, 4, GF_BN);

    // layer residual
    const float* Rt = nullptr;
    if (c.res) {
      if (cin == cout && s == 1) {
        Rt = Xc;
      } else {
        dim3 gRT((Mo + 63) / 64, cout / 16, 64);
        gemm_conv_kernel<1><<<gRT, 128, 0, stream>>>(
            Xc, l.resTcn.w, l.resTcn.b,
            l.resTcnBn.gamma, l.resTcnBn.beta, l.resTcnBn.mean, l.resTcnBn.var,
            bufR, cout, cin, T, Tout, s, 0, GF_BN);
        Rt = bufR;
      }
    }
    const unsigned totT = (unsigned)(64u * cout * Mo);
    add_relu_kernel<<<(totT + 255) / 256, 256, 0, stream>>>(Yn, Rt, totT);

    cur = 1 - cur;
    T = Tout;
  }

  // ---- final: mean-pool over (M=2, T*V) then FC to 60 classes ----
  pool_kernel<<<32 * 256, 128, 0, stream>>>(bufX[cur], poolBuf, 256, T * 25);
  fc_kernel<<<32, 64, 0, stream>>>(poolBuf, fc_w, fc_b, (float*)d_out, 256);
}
